// GAMLP_learnable_emb_Module_38242388803876
// MI455X (gfx1250) — compile-verified, matrix operand order checked by reading, not run on hardware
//
#include <hip/hip_runtime.h>
#include <hip/hip_bf16.h>
#include <math.h>

// ---------------------------------------------------------------------------
// GAMLP forward for MI455X (gfx1250).
// GEMMs run on v_wmma_f32_16x16x32_bf16 (bf16 in, f32 accumulate) with 32x64
// register tiles per wave and fused FFNII epilogues; SpMM hops use f32
// hardware atomics; attention is a wave-per-node reduction kernel.
// Needs ~590 MB of d_ws.
// ---------------------------------------------------------------------------

#define NN 100000      // nodes
#define EE 500000      // edges
#define DD 128         // emb dim
#define HH 512         // hidden
typedef unsigned short u16;

typedef __attribute__((ext_vector_type(16))) __bf16 v16bf;
typedef __attribute__((ext_vector_type(8)))  float  v8f;
typedef __attribute__((ext_vector_type(8)))  u16    v8u;

__device__ __forceinline__ u16 f2bf(float f) {
    unsigned int u = __float_as_uint(f);
    unsigned int r = u + 0x7FFFu + ((u >> 16) & 1u);   // round-to-nearest-even
    return (u16)(r >> 16);
}
__device__ __forceinline__ float bf2f(u16 h) {
    return __uint_as_float(((unsigned int)h) << 16);
}

// Load a 16-element bf16 WMMA fragment: two contiguous 16-byte loads
// (elements k..k+7 and k+16..k+23 relative to this lane's k-base).
__device__ __forceinline__ v16bf ldfrag(const u16* p) {
    union { struct { v8u lo; v8u hi; } s; v16bf v; } u;
    u.s.lo = *(const v8u*)(p);
    u.s.hi = *(const v8u*)(p + 16);
    return u.v;
}

// ---------------------------------------------------------------------------
// Weight convert+transpose:  W [K,M] f32 row-major  ->  Wt [M,K] bf16
// ---------------------------------------------------------------------------
__global__ void wconvert(const float* __restrict__ W, u16* __restrict__ Wt,
                         int K, int M) {
    long i = (long)blockIdx.x * blockDim.x + threadIdx.x;
    long total = (long)K * M;
    if (i >= total) return;
    int k = (int)(i / M), m = (int)(i % M);
    Wt[(size_t)m * K + k] = f2bf(W[i]);
}

// f32 [NN,DD] -> bf16 columns [colOff .. colOff+DD) of concat [NN, 4*DD]
__global__ void cast_cols(const float* __restrict__ src, u16* __restrict__ concat,
                          int colOff) {
    long i = (long)blockIdx.x * blockDim.x + threadIdx.x;
    if (i >= (long)NN * DD) return;
    long node = i / DD; int d = (int)(i % DD);
    concat[node * (4 * DD) + colOff + d] = f2bf(src[i]);
}

__global__ void zero_f32(float* p, long n) {
    long i = (long)blockIdx.x * blockDim.x + threadIdx.x;
    if (i < n) p[i] = 0.0f;
}

// ---------------------------------------------------------------------------
// SpMM hop:  acc[dst[e]] += h[src[e]] * ew[e]   (one wave per edge, f32 atomics)
// ---------------------------------------------------------------------------
__global__ void spmm_hop(const float* __restrict__ h, const int* __restrict__ src,
                         const int* __restrict__ dst, const float* __restrict__ ew,
                         float* __restrict__ acc) {
    long t = (long)blockIdx.x * blockDim.x + threadIdx.x;
    long e = t >> 5;
    if (e >= EE) return;
    int lane = (int)(t & 31);
    int s = src[e], d = dst[e];
    float w = ew[e];
    const float4 v = *(const float4*)(h + (size_t)s * DD + lane * 4);
    float* ap = acc + (size_t)d * DD + lane * 4;
    unsafeAtomicAdd(ap + 0, v.x * w);
    unsafeAtomicAdd(ap + 1, v.y * w);
    unsafeAtomicAdd(ap + 2, v.z * w);
    unsafeAtomicAdd(ap + 3, v.w * w);
}

// ---------------------------------------------------------------------------
// bf16 WMMA GEMM:  out = A[N,K] @ B[K,M]  (B supplied transposed as Bt[M,K])
// Each wave: 32 rows x 64 cols (2 A frags x 4 B frags -> 8 WMMA / K-step).
// Fused FFNII epilogues:
//   mode 0: f_out = acc + bias
//   mode 1: h0_out = bf16(acc+bias); bf_out = bf16(0.5*(prelu(v)+v))
//   mode 2: bf_out = bf16(0.5*(prelu(acc) + h0_in))
//   mode 3: bf_out = bf16(prelu(acc))
//   mode 4: f_out = prelu(acc + bias)          (outer PReLU on jk)
// ---------------------------------------------------------------------------
__global__ void __launch_bounds__(256)
gemm_bf16_wmma(const u16* __restrict__ A, const u16* __restrict__ Bt,
               int Nrows, int K, int M,
               const float* __restrict__ bias, const float* __restrict__ slope,
               const u16* __restrict__ h0_in,
               u16* __restrict__ h0_out, u16* __restrict__ bf_out,
               float* __restrict__ f_out, int mode) {
    const int wave = threadIdx.x >> 5;
    const int lane = threadIdx.x & 31;
    const int rowBase = (blockIdx.x * 8 + wave) * 32;     // 32-row strip per wave
    if (rowBase >= Nrows) return;              // wave-uniform: EXEC stays all-1s
    const int colBase = blockIdx.y * 64;
    const int m = lane & 15;
    const int hi = lane >> 4;

    const u16* arow0 = A  + (size_t)(rowBase + m) * K + hi * 8;        // rows 0..15
    const u16* arow1 = arow0 + (size_t)16 * K;                          // rows 16..31
    const u16* bcol  = Bt + (size_t)(colBase + m) * K + hi * 8;

    v8f acc[8] = {};
    for (int kb = 0; kb < K; kb += 32) {
        __builtin_prefetch(arow0 + kb + 64, 0, 1);   // global_prefetch_b8
        v16bf a0 = ldfrag(arow0 + kb);
        v16bf a1 = ldfrag(arow1 + kb);
        v16bf bf[4];
        bf[0] = ldfrag(bcol + kb);
        bf[1] = ldfrag(bcol + (size_t)16 * K + kb);
        bf[2] = ldfrag(bcol + (size_t)32 * K + kb);
        bf[3] = ldfrag(bcol + (size_t)48 * K + kb);
#pragma unroll
        for (int c = 0; c < 4; ++c) {
            acc[c]     = __builtin_amdgcn_wmma_f32_16x16x32_bf16(
                false, a0, false, bf[c], (short)0, acc[c], false, false);
            acc[4 + c] = __builtin_amdgcn_wmma_f32_16x16x32_bf16(
                false, a1, false, bf[c], (short)0, acc[4 + c], false, false);
        }
    }

    const float a = slope ? slope[0] : 0.0f;
#pragma unroll
    for (int rg = 0; rg < 2; ++rg) {
#pragma unroll
        for (int c = 0; c < 4; ++c) {
            const int col = colBase + c * 16 + m;
            const float bv = bias ? bias[col] : 0.0f;
#pragma unroll
            for (int j = 0; j < 8; ++j) {
                const int r = rowBase + rg * 16 + (hi << 3) + j;
                const size_t idx = (size_t)r * M + col;
                float v = acc[rg * 4 + c][j] + bv;
                if (mode == 0) {
                    f_out[idx] = v;
                } else if (mode == 4) {
                    f_out[idx] = v > 0.0f ? v : a * v;
                } else if (mode == 1) {
                    h0_out[idx] = f2bf(v);
                    const float pr = v > 0.0f ? v : a * v;
                    bf_out[idx] = f2bf(0.5f * (pr + v));          // ALPHA = 0.5
                } else if (mode == 2) {
                    const float h0 = bf2f(h0_in[idx]);
                    const float pr = v > 0.0f ? v : a * v;
                    bf_out[idx] = f2bf(0.5f * (pr + h0));
                } else {
                    const float pr = v > 0.0f ? v : a * v;
                    bf_out[idx] = f2bf(pr);
                }
            }
        }
    }
}

// ---------------------------------------------------------------------------
// Hop attention: scores_i = tanh(jk.Wt + f_i.Wb + b), softmax over 4 hops,
// right = sum_i w_i * f_i.  One wave per node; butterfly reductions.
// ---------------------------------------------------------------------------
__global__ void __launch_bounds__(256)
attention_kernel(const float* __restrict__ jk, const u16* __restrict__ concat,
                 const float* __restrict__ Watt, const float* __restrict__ batt,
                 u16* __restrict__ right_bf) {
    long t = (long)blockIdx.x * blockDim.x + threadIdx.x;
    long node = t >> 5;
    if (node >= NN) return;
    int lane = (int)(t & 31);
    int d0 = lane * 4;

    float jd = 0.0f, wb[4];
#pragma unroll
    for (int i = 0; i < 4; ++i) {
        jd += jk[node * DD + d0 + i] * Watt[d0 + i];
        wb[i] = Watt[DD + d0 + i];
    }
#pragma unroll
    for (int o = 16; o > 0; o >>= 1) jd += __shfl_xor(jd, o, 32);

    float f[4][4], sc[4];
    const float bb = batt[0];
#pragma unroll
    for (int hop = 0; hop < 4; ++hop) {
        float fd = 0.0f;
#pragma unroll
        for (int i = 0; i < 4; ++i) {
            f[hop][i] = bf2f(concat[node * (4 * DD) + hop * DD + d0 + i]);
            fd += f[hop][i] * wb[i];
        }
#pragma unroll
        for (int o = 16; o > 0; o >>= 1) fd += __shfl_xor(fd, o, 32);
        sc[hop] = tanhf(jd + fd + bb);
    }
    float mx = fmaxf(fmaxf(sc[0], sc[1]), fmaxf(sc[2], sc[3]));
    float e0 = expf(sc[0] - mx), e1 = expf(sc[1] - mx);
    float e2 = expf(sc[2] - mx), e3 = expf(sc[3] - mx);
    float inv = 1.0f / (e0 + e1 + e2 + e3);
    float w0 = e0 * inv, w1 = e1 * inv, w2 = e2 * inv, w3 = e3 * inv;
#pragma unroll
    for (int i = 0; i < 4; ++i) {
        float r = w0 * f[0][i] + w1 * f[1][i] + w2 * f[2][i] + w3 * f[3][i];
        right_bf[node * DD + d0 + i] = f2bf(r);
    }
}

// ---------------------------------------------------------------------------
extern "C" void kernel_launch(void* const* d_in, const int* in_sizes, int n_in,
                              void* d_out, int out_size, void* d_ws, size_t ws_size,
                              hipStream_t stream) {
    if (n_in < 21) return;
    const float* x    = (const float*)d_in[0];
    const int*   srcI = (const int*)  d_in[1];
    const int*   dstI = (const int*)  d_in[2];
    const float* ew   = (const float*)d_in[3];
    const float* jkW0 = (const float*)d_in[4];
    const float* jkb0 = (const float*)d_in[5];
    const float* jkW1 = (const float*)d_in[6];
    const float* jkW2 = (const float*)d_in[7];
    const float* jkW3 = (const float*)d_in[8];
    const float* jkb3 = (const float*)d_in[9];
    const float* jka  = (const float*)d_in[10];
    const float* ajk  = (const float*)d_in[11];
    const float* Watt = (const float*)d_in[12];
    const float* batt = (const float*)d_in[13];
    const float* oW0  = (const float*)d_in[14];
    const float* ob0  = (const float*)d_in[15];
    const float* oW1  = (const float*)d_in[16];
    const float* oW2  = (const float*)d_in[17];
    const float* oW3  = (const float*)d_in[18];
    const float* ob3  = (const float*)d_in[19];
    const float* oa   = (const float*)d_in[20];
    float* out = (float*)d_out;

    // -------- workspace carve-up --------
    char* w = (char*)d_ws;
    size_t off = 0;
    auto take = [&](size_t bytes) -> char* {
        char* p = w + off;
        off = (off + bytes + 255) & ~(size_t)255;
        return p;
    };
    u16*   concat = (u16*)  take((size_t)NN * 4 * DD * 2);   // bf16 [N, 512]
    float* accA   = (float*)take((size_t)NN * DD * 4);
    float* accB   = (float*)take((size_t)NN * DD * 4);
    u16*   h0b    = (u16*)  take((size_t)NN * HH * 2);
    u16*   hinA   = (u16*)  take((size_t)NN * HH * 2);
    u16*   hinB   = (u16*)  take((size_t)NN * HH * 2);
    float* jkf    = (float*)take((size_t)NN * DD * 4);
    u16*   rightb = (u16*)  take((size_t)NN * DD * 2);
    u16* WtJ0 = (u16*)take((size_t)HH * HH * 2);   // [512,512]^T
    u16* WtJ1 = (u16*)take((size_t)HH * HH * 2);
    u16* WtJ2 = (u16*)take((size_t)HH * HH * 2);
    u16* WtJ3 = (u16*)take((size_t)HH * DD * 2);   // [512,128]^T
    u16* WtO0 = (u16*)take((size_t)DD * HH * 2);   // [128,512]^T
    u16* WtO1 = (u16*)take((size_t)HH * HH * 2);
    u16* WtO2 = (u16*)take((size_t)HH * HH * 2);
    u16* WtO3 = (u16*)take((size_t)HH * DD * 2);
    (void)ws_size; (void)in_sizes; (void)out_size;

    // -------- weight conversion (f32 -> bf16, transposed) --------
    auto wconv = [&](const float* W, u16* Wt, int K, int M) {
        long total = (long)K * M;
        wconvert<<<dim3((unsigned)((total + 255) / 256)), dim3(256), 0, stream>>>(W, Wt, K, M);
    };
    wconv(jkW0, WtJ0, HH, HH);  wconv(jkW1, WtJ1, HH, HH);
    wconv(jkW2, WtJ2, HH, HH);  wconv(jkW3, WtJ3, HH, DD);
    wconv(oW0,  WtO0, DD, HH);  wconv(oW1,  WtO1, HH, HH);
    wconv(oW2,  WtO2, HH, HH);  wconv(oW3,  WtO3, HH, DD);

    // -------- hop propagation --------
    const long nd = (long)NN * DD;
    const unsigned ndBlocks  = (unsigned)((nd + 255) / 256);
    const unsigned edgeBlocks = (unsigned)(((long)EE * 32 + 255) / 256);
    cast_cols<<<ndBlocks, 256, 0, stream>>>(x, concat, 0);
    zero_f32<<<ndBlocks, 256, 0, stream>>>(accA, nd);
    spmm_hop<<<edgeBlocks, 256, 0, stream>>>(x, srcI, dstI, ew, accA);
    cast_cols<<<ndBlocks, 256, 0, stream>>>(accA, concat, DD);
    zero_f32<<<ndBlocks, 256, 0, stream>>>(accB, nd);
    spmm_hop<<<edgeBlocks, 256, 0, stream>>>(accA, srcI, dstI, ew, accB);
    cast_cols<<<ndBlocks, 256, 0, stream>>>(accB, concat, 2 * DD);
    zero_f32<<<ndBlocks, 256, 0, stream>>>(accA, nd);
    spmm_hop<<<edgeBlocks, 256, 0, stream>>>(accB, srcI, dstI, ew, accA);
    cast_cols<<<ndBlocks, 256, 0, stream>>>(accA, concat, 3 * DD);

    // -------- GEMM helper --------
    const unsigned rowBlocks = (3125 + 7) / 8;   // NN/32 row strips, 8 per block
    auto gemm = [&](const u16* A, const u16* Bt, int K, int M,
                    const float* bias, const float* slope, const u16* h0in,
                    u16* h0out, u16* bfout, float* fout, int mode) {
        dim3 grid(rowBlocks, (unsigned)(M / 64));
        gemm_bf16_wmma<<<grid, dim3(256), 0, stream>>>(
            A, Bt, NN, K, M, bias, slope, h0in, h0out, bfout, fout, mode);
    };

    // -------- jk = PReLU(FFNII(concat), a_jk) --------
    gemm(concat, WtJ0, 4 * DD, HH, jkb0, jka, nullptr, h0b, hinA, nullptr, 1);
    gemm(hinA,   WtJ1, HH,     HH, nullptr, jka, h0b, nullptr, hinB, nullptr, 2);
    gemm(hinB,   WtJ2, HH,     HH, nullptr, jka, nullptr, nullptr, hinA, nullptr, 3);
    gemm(hinA,   WtJ3, HH,     DD, jkb3, ajk, nullptr, nullptr, nullptr, jkf, 4);

    // -------- hop attention + weighted sum --------
    attention_kernel<<<(unsigned)(((long)NN * 32 + 255) / 256), 256, 0, stream>>>(
        jkf, concat, Watt, batt, rightb);

    // -------- output FFNII(right) -> d_out --------
    gemm(rightb, WtO0, DD, HH, ob0, oa, nullptr, h0b, hinA, nullptr, 1);
    gemm(hinA,   WtO1, HH, HH, nullptr, oa, h0b, nullptr, hinB, nullptr, 2);
    gemm(hinB,   WtO2, HH, HH, nullptr, oa, nullptr, nullptr, hinA, nullptr, 3);
    gemm(hinA,   WtO3, HH, DD, ob3, nullptr, nullptr, nullptr, nullptr, out, 0);
}
